// HyperGraphAttentionLayer_58171037057642
// MI455X (gfx1250) — matrix operand order repaired and checked by reading, not compiled, not run on hardware
//
#include <hip/hip_runtime.h>
#include <hip/hip_bf16.h>
#include <stddef.h>

// ---------------------------------------------------------------------------
// HyperGraph attention, MI455X (gfx1250, wave32, WMMA bf16 16x16x32).
// Shapes fixed by the reference: B=8, N=4096, E=2048, D=256.
// Strategy: pack ht into a bitmask (8.4MB, L2-resident); bf16 WMMA for all
// contractions with fp32 accumulation; flash-style two-pass softmax for both
// stages (stats pass + recompute/PV pass with P staged in LDS).
// ---------------------------------------------------------------------------

constexpr int gB = 8, gN = 4096, gE = 2048, gD = 256;
constexpr float kScale = 0.0625f;          // 1/sqrt(256)
constexpr float kNegBig = -9.0e15f;        // INF_SUB_NUM from reference
constexpr int gNW = gN / 32;               // ht bitmask words per edge row

typedef __attribute__((ext_vector_type(16))) __bf16 v16bf;
typedef __attribute__((ext_vector_type(8)))  float  v8f;

// ---- WMMA fragment helpers (16x16x32 bf16; A and B share contiguous k = i + 16h) -------

__device__ __forceinline__ v8f wmma_bf16(v16bf a, v16bf b, v8f c) {
  return __builtin_amdgcn_wmma_f32_16x16x32_bf16(
      false, a, false, b, (short)0, c, false, false);
}

// A matrix 16(M)x32(K): lane l holds row M=l&15; K = i + (l>=16?16:0)
__device__ __forceinline__ v16bf load_a_frag(const __bf16* base, int ld) {
  const int lane = threadIdx.x & 31;
  const __bf16* row = base + (size_t)(lane & 15) * ld + ((lane >> 4) << 4);
  v16bf a;
#pragma unroll
  for (int i = 0; i < 16; ++i) a[i] = row[i];
  return a;
}

// B matrix 32(K)x16(N) loaded from a [N][K]-major source (row = column of B):
// lane l holds column N=l&15; K = i + (l>=16?16:0)  (contiguous 32B per lane)
__device__ __forceinline__ v16bf load_b_frag_rows(const __bf16* base, int ld) {
  const int lane = threadIdx.x & 31;
  const __bf16* row = base + (size_t)(lane & 15) * ld + ((lane >> 4) << 4);
  v16bf b;
#pragma unroll
  for (int i = 0; i < 16; ++i) b[i] = row[i];
  return b;
}

// C/D 16x16 f32: lane l holds column N=l&15; element j -> row j + 8*(l>=16)
__device__ __forceinline__ void store_c_f32(float* base, int ld, v8f c) {
  const int lane = threadIdx.x & 31;
  float* col = base + (lane & 15);
  const int r0 = (lane >> 4) << 3;
#pragma unroll
  for (int j = 0; j < 8; ++j) col[(size_t)(r0 + j) * ld] = c[j];
}

__device__ __forceinline__ void store_c_bf(__bf16* base, int ld, v8f c) {
  const int lane = threadIdx.x & 31;
  __bf16* col = base + (lane & 15);
  const int r0 = (lane >> 4) << 3;
#pragma unroll
  for (int j = 0; j < 8; ++j) col[(size_t)(r0 + j) * ld] = (__bf16)c[j];
}

// ---- prep kernels ----------------------------------------------------------

__global__ void k_layernorm(const float* __restrict__ x, const float* __restrict__ g,
                            const float* __restrict__ bt, float* __restrict__ y,
                            __bf16* __restrict__ ybf) {
  __shared__ float red[gD];
  const int row = blockIdx.x;
  const int t = threadIdx.x;
  const float v = x[(size_t)row * gD + t];
  red[t] = v; __syncthreads();
  for (int s = gD / 2; s > 0; s >>= 1) { if (t < s) red[t] += red[t + s]; __syncthreads(); }
  const float mu = red[0] * (1.0f / gD);
  __syncthreads();
  const float dv = v - mu;
  red[t] = dv * dv; __syncthreads();
  for (int s = gD / 2; s > 0; s >>= 1) { if (t < s) red[t] += red[t + s]; __syncthreads(); }
  const float var = red[0] * (1.0f / gD);
  const float o = g[t] * dv * rsqrtf(var + 1e-5f) + bt[t];
  y[(size_t)row * gD + t] = o;
  ybf[(size_t)row * gD + t] = (__bf16)o;
}

__global__ void k_cvt_hidden(const float* __restrict__ h, __bf16* __restrict__ hb,
                             __bf16* __restrict__ hT) {
  const size_t i = (size_t)blockIdx.x * blockDim.x + threadIdx.x;  // over B*N*D
  const int d = (int)(i % gD);
  const size_t r = i / gD;
  const int n = (int)(r % gN);
  const int b = (int)(r / gN);
  const __bf16 v = (__bf16)h[i];
  hb[i] = v;
  hT[((size_t)b * gD + d) * gN + n] = v;
}

__global__ void k_cvt_wT(const float* __restrict__ w, __bf16* __restrict__ wT) {
  const int i = blockIdx.x * blockDim.x + threadIdx.x;   // over D*D
  const int r = i / gD, c = i % gD;
  wT[(size_t)c * gD + r] = (__bf16)w[i];
}

__global__ void k_pack_ht(const int* __restrict__ ht, unsigned* __restrict__ htb) {
  const size_t i = (size_t)blockIdx.x * blockDim.x + threadIdx.x; // word index
  const int* p = ht + i * 32;
  unsigned wv = 0;
#pragma unroll
  for (int k = 0; k < 32; ++k) wv |= (p[k] > 0 ? 1u : 0u) << k;
  htb[i] = wv;
}

// ---- generic row-tile GEMM: C_bf[M,256] = A_bf[M,256] @ W (given as W^T) ----

__global__ void k_gemm_bf(const __bf16* __restrict__ A, const __bf16* __restrict__ WT,
                          __bf16* __restrict__ C) {
  const int mt = blockIdx.x * 16;
  const int w = threadIdx.x >> 5;
  const int nb = w * 64;
  v8f acc0 = {}, acc1 = {}, acc2 = {}, acc3 = {};
#pragma unroll
  for (int kk = 0; kk < 8; ++kk) {
    const v16bf a = load_a_frag(A + (size_t)mt * gD + kk * 32, gD);
    acc0 = wmma_bf16(a, load_b_frag_rows(WT + (size_t)(nb +  0) * gD + kk * 32, gD), acc0);
    acc1 = wmma_bf16(a, load_b_frag_rows(WT + (size_t)(nb + 16) * gD + kk * 32, gD), acc1);
    acc2 = wmma_bf16(a, load_b_frag_rows(WT + (size_t)(nb + 32) * gD + kk * 32, gD), acc2);
    acc3 = wmma_bf16(a, load_b_frag_rows(WT + (size_t)(nb + 48) * gD + kk * 32, gD), acc3);
  }
  store_c_bf(C + (size_t)mt * gD + nb +  0, gD, acc0);
  store_c_bf(C + (size_t)mt * gD + nb + 16, gD, acc1);
  store_c_bf(C + (size_t)mt * gD + nb + 32, gD, acc2);
  store_c_bf(C + (size_t)mt * gD + nb + 48, gD, acc3);
}

// ---- stage 1 pass A: per-(b,e) max / sumexp over n --------------------------

__global__ void k_edge_stats(const __bf16* __restrict__ lnEbf,
                             const __bf16* __restrict__ nodek,
                             const unsigned* __restrict__ htb,
                             float* __restrict__ m1, float* __restrict__ l1) {
  const int b = blockIdx.y;
  const int et = blockIdx.x * 16;
  const int w = threadIdx.x >> 5;
  const int lane = threadIdx.x & 31;
  const int col = lane & 15;
  const int r0 = (lane >> 4) << 3;

  v16bf afr[8];
  const __bf16* Ab = lnEbf + (size_t)(b * gE + et) * gD;
#pragma unroll
  for (int kk = 0; kk < 8; ++kk) afr[kk] = load_a_frag(Ab + kk * 32, gD);

  float M[8], L[8];
#pragma unroll
  for (int j = 0; j < 8; ++j) { M[j] = -3.0e38f; L[j] = 0.0f; }

  for (int nb = w * 16; nb < gN; nb += 64) {
    v8f c = {};
    const __bf16* Bb = nodek + (size_t)(b * gN + nb) * gD;
#pragma unroll
    for (int kk = 0; kk < 8; ++kk)
      c = wmma_bf16(afr[kk], load_b_frag_rows(Bb + kk * 32, gD), c);
    const int n = nb + col;
    const int wn = n >> 5, bit = n & 31;
    float s[8];
#pragma unroll
    for (int j = 0; j < 8; ++j) {
      const int e = et + r0 + j;
      const unsigned mw = htb[(size_t)(b * gE + e) * gNW + wn];
      s[j] = ((mw >> bit) & 1u) ? c[j] * kScale : kNegBig;
    }
#pragma unroll
    for (int j = 0; j < 8; ++j) {           // per-row online update (16 cols/half)
      float cm = s[j];
      cm = fmaxf(cm, __shfl_xor(cm, 1, 32));
      cm = fmaxf(cm, __shfl_xor(cm, 2, 32));
      cm = fmaxf(cm, __shfl_xor(cm, 4, 32));
      cm = fmaxf(cm, __shfl_xor(cm, 8, 32));
      const float nM = fmaxf(M[j], cm);
      float es = __expf(s[j] - nM);
      es += __shfl_xor(es, 1, 32);
      es += __shfl_xor(es, 2, 32);
      es += __shfl_xor(es, 4, 32);
      es += __shfl_xor(es, 8, 32);
      L[j] = L[j] * __expf(M[j] - nM) + es;
      M[j] = nM;
    }
  }
  __shared__ float smM[4][16], smL[4][16];
  if (col == 0) {
#pragma unroll
    for (int j = 0; j < 8; ++j) { smM[w][r0 + j] = M[j]; smL[w][r0 + j] = L[j]; }
  }
  __syncthreads();
  if (threadIdx.x < 16) {
    const int r = threadIdx.x;
    float Mf = -3.0e38f;
    for (int ww = 0; ww < 4; ++ww) Mf = fmaxf(Mf, smM[ww][r]);
    float Lf = 0.0f;
    for (int ww = 0; ww < 4; ++ww) Lf += smL[ww][r] * __expf(smM[ww][r] - Mf);
    m1[(size_t)b * gE + et + r] = Mf;
    l1[(size_t)b * gE + et + r] = Lf;
  }
}

// ---- stage 1 pass B: edge_h = softmax(S) @ hidden + lnE ---------------------

__global__ void k_edge_h(const __bf16* __restrict__ lnEbf, const float* __restrict__ lnEf,
                         const __bf16* __restrict__ nodek, const __bf16* __restrict__ hidT,
                         const unsigned* __restrict__ htb,
                         const float* __restrict__ m1, const float* __restrict__ l1,
                         __bf16* __restrict__ ehbf, __bf16* __restrict__ ehT) {
  const int b = blockIdx.y;
  const int et = blockIdx.x * 16;
  const int w = threadIdx.x >> 5;
  const int lane = threadIdx.x & 31;
  const int col = lane & 15;
  const int r0 = (lane >> 4) << 3;

  __shared__ __bf16 P[16 * 64];

  v16bf afr[8];
  const __bf16* Ab = lnEbf + (size_t)(b * gE + et) * gD;
#pragma unroll
  for (int kk = 0; kk < 8; ++kk) afr[kk] = load_a_frag(Ab + kk * 32, gD);

  float rM[8], rInvL[8];
#pragma unroll
  for (int j = 0; j < 8; ++j) {
    rM[j] = m1[(size_t)b * gE + et + r0 + j];
    rInvL[j] = 1.0f / l1[(size_t)b * gE + et + r0 + j];
  }

  v8f acc0 = {}, acc1 = {}, acc2 = {}, acc3 = {};
  for (int nb = 0; nb < gN; nb += 64) {
    // scores for this wave's 16 nodes
    v8f c = {};
    const __bf16* Bb = nodek + (size_t)(b * gN + nb + w * 16) * gD;
#pragma unroll
    for (int kk = 0; kk < 8; ++kk)
      c = wmma_bf16(afr[kk], load_b_frag_rows(Bb + kk * 32, gD), c);
    const int n = nb + w * 16 + col;
    const int wn = n >> 5, bit = n & 31;
#pragma unroll
    for (int j = 0; j < 8; ++j) {
      const int e = et + r0 + j;
      const unsigned mw = htb[(size_t)(b * gE + e) * gNW + wn];
      const float s = ((mw >> bit) & 1u) ? c[j] * kScale : kNegBig;
      const float p = __expf(s - rM[j]) * rInvL[j];
      P[(r0 + j) * 64 + w * 16 + col] = (__bf16)p;
    }
    __syncthreads();
    // PV: this wave's 64 output columns, K = 64 nodes of this chunk
#pragma unroll
    for (int kk2 = 0; kk2 < 2; ++kk2) {
      const v16bf ap = load_a_frag((const __bf16*)P + kk2 * 32, 64);
      acc0 = wmma_bf16(ap, load_b_frag_rows(hidT + (size_t)(b * gD + w * 64 +  0) * gN + nb + kk2 * 32, gN), acc0);
      acc1 = wmma_bf16(ap, load_b_frag_rows(hidT + (size_t)(b * gD + w * 64 + 16) * gN + nb + kk2 * 32, gN), acc1);
      acc2 = wmma_bf16(ap, load_b_frag_rows(hidT + (size_t)(b * gD + w * 64 + 32) * gN + nb + kk2 * 32, gN), acc2);
      acc3 = wmma_bf16(ap, load_b_frag_rows(hidT + (size_t)(b * gD + w * 64 + 48) * gN + nb + kk2 * 32, gN), acc3);
    }
    __syncthreads();
  }
  // add lnE residual, store row-major bf16 and transposed bf16
  v8f accs[4] = {acc0, acc1, acc2, acc3};
#pragma unroll
  for (int t = 0; t < 4; ++t) {
    const int dbase = w * 64 + t * 16;
#pragma unroll
    for (int j = 0; j < 8; ++j) {
      const int e = et + r0 + j;
      const int d = dbase + col;
      const float v = accs[t][j] + lnEf[(size_t)(b * gE + e) * gD + d];
      ehbf[(size_t)(b * gE + e) * gD + d] = (__bf16)v;
      ehT[((size_t)b * gD + d) * gE + e] = (__bf16)v;
    }
  }
}

// ---- stage 2 pass A: per-(b,e) max / sumexp over n (column stats) -----------

__global__ void k_node_stats(const __bf16* __restrict__ nodek,
                             const __bf16* __restrict__ edgek,
                             const unsigned* __restrict__ htb,
                             float* __restrict__ m2, float* __restrict__ l2) {
  const int b = blockIdx.y;
  const int et = blockIdx.x * 16;
  const int w = threadIdx.x >> 5;
  const int lane = threadIdx.x & 31;
  const int col = lane & 15;
  const int r0 = (lane >> 4) << 3;

  v16bf bfr[8];
  const __bf16* Bb = edgek + (size_t)(b * gE + et) * gD;
#pragma unroll
  for (int kk = 0; kk < 8; ++kk) bfr[kk] = load_b_frag_rows(Bb + kk * 32, gD);

  const int e = et + col;
  const size_t htrow = (size_t)(b * gE + e) * gNW;
  float Mc = -3.0e38f, Lc = 0.0f;

  for (int nt = w * 16; nt < gN; nt += 64) {
    v8f c = {};
    const __bf16* Ab = nodek + (size_t)(b * gN + nt) * gD;
#pragma unroll
    for (int kk = 0; kk < 8; ++kk)
      c = wmma_bf16(load_a_frag(Ab + kk * 32, gD), bfr[kk], c);
    float s[8];
#pragma unroll
    for (int j = 0; j < 8; ++j) {
      const int n = nt + r0 + j;
      const unsigned mw = htb[htrow + (n >> 5)];
      s[j] = ((mw >> (n & 31)) & 1u) ? c[j] * kScale : kNegBig;
    }
    float cm = s[0];
#pragma unroll
    for (int j = 1; j < 8; ++j) cm = fmaxf(cm, s[j]);
    cm = fmaxf(cm, __shfl_xor(cm, 16, 32));      // combine row-halves (same col)
    const float nM = fmaxf(Mc, cm);
    float es = 0.0f;
#pragma unroll
    for (int j = 0; j < 8; ++j) es += __expf(s[j] - nM);
    es += __shfl_xor(es, 16, 32);
    Lc = Lc * __expf(Mc - nM) + es;
    Mc = nM;
  }
  __shared__ float smM[4][16], smL[4][16];
  if (lane < 16) { smM[w][col] = Mc; smL[w][col] = Lc; }
  __syncthreads();
  if (threadIdx.x < 16) {
    const int r = threadIdx.x;
    float Mf = -3.0e38f;
    for (int ww = 0; ww < 4; ++ww) Mf = fmaxf(Mf, smM[ww][r]);
    float Lf = 0.0f;
    for (int ww = 0; ww < 4; ++ww) Lf += smL[ww][r] * __expf(smM[ww][r] - Mf);
    m2[(size_t)b * gE + et + r] = Mf;
    l2[(size_t)b * gE + et + r] = Lf;
  }
}

// ---- stage 2 pass B: node_hidden = softmax_col(S2) @ edge_h -----------------

__global__ void k_node_out(const __bf16* __restrict__ nodek, const __bf16* __restrict__ edgek,
                           const __bf16* __restrict__ ehT, const unsigned* __restrict__ htb,
                           const float* __restrict__ m2, const float* __restrict__ l2,
                           float* __restrict__ out) {
  const int b = blockIdx.y;
  const int nt = blockIdx.x * 16;
  const int w = threadIdx.x >> 5;
  const int lane = threadIdx.x & 31;
  const int col = lane & 15;
  const int r0 = (lane >> 4) << 3;

  __shared__ __bf16 P[16 * 64];

  v16bf afr[8];
  const __bf16* Ab = nodek + (size_t)(b * gN + nt) * gD;
#pragma unroll
  for (int kk = 0; kk < 8; ++kk) afr[kk] = load_a_frag(Ab + kk * 32, gD);

  v8f acc0 = {}, acc1 = {}, acc2 = {}, acc3 = {};
  for (int eb = 0; eb < gE; eb += 64) {
    v8f c = {};
    const __bf16* Bb = edgek + (size_t)(b * gE + eb + w * 16) * gD;
#pragma unroll
    for (int kk = 0; kk < 8; ++kk)
      c = wmma_bf16(afr[kk], load_b_frag_rows(Bb + kk * 32, gD), c);
    const int e = eb + w * 16 + col;
    const float Me = m2[(size_t)b * gE + e];
    const float invL = 1.0f / l2[(size_t)b * gE + e];
    const size_t htrow = (size_t)(b * gE + e) * gNW;
#pragma unroll
    for (int j = 0; j < 8; ++j) {
      const int n = nt + r0 + j;
      const unsigned mw = htb[htrow + (n >> 5)];
      const float s = ((mw >> (n & 31)) & 1u) ? c[j] * kScale : kNegBig;
      P[(r0 + j) * 64 + w * 16 + col] = (__bf16)(__expf(s - Me) * invL);
    }
    __syncthreads();
#pragma unroll
    for (int kk2 = 0; kk2 < 2; ++kk2) {
      const v16bf ap = load_a_frag((const __bf16*)P + kk2 * 32, 64);
      acc0 = wmma_bf16(ap, load_b_frag_rows(ehT + (size_t)(b * gD + w * 64 +  0) * gE + eb + kk2 * 32, gE), acc0);
      acc1 = wmma_bf16(ap, load_b_frag_rows(ehT + (size_t)(b * gD + w * 64 + 16) * gE + eb + kk2 * 32, gE), acc1);
      acc2 = wmma_bf16(ap, load_b_frag_rows(ehT + (size_t)(b * gD + w * 64 + 32) * gE + eb + kk2 * 32, gE), acc2);
      acc3 = wmma_bf16(ap, load_b_frag_rows(ehT + (size_t)(b * gD + w * 64 + 48) * gE + eb + kk2 * 32, gE), acc3);
    }
    __syncthreads();
  }
  store_c_f32(out + (size_t)(b * gN + nt) * gD + w * 64 +  0, gD, acc0);
  store_c_f32(out + (size_t)(b * gN + nt) * gD + w * 64 + 16, gD, acc1);
  store_c_f32(out + (size_t)(b * gN + nt) * gD + w * 64 + 32, gD, acc2);
  store_c_f32(out + (size_t)(b * gN + nt) * gD + w * 64 + 48, gD, acc3);
}

// ---------------------------------------------------------------------------

extern "C" void kernel_launch(void* const* d_in, const int* in_sizes, int n_in,
                              void* d_out, int out_size, void* d_ws, size_t ws_size,
                              hipStream_t stream) {
  (void)in_sizes; (void)n_in; (void)out_size; (void)ws_size;
  const float* hidden   = (const float*)d_in[0];
  const int*   ht       = (const int*)d_in[1];
  const float* edge_emb = (const float*)d_in[2];
  const float* wnk      = (const float*)d_in[3];
  const float* wek      = (const float*)d_in[4];
  const float* lng      = (const float*)d_in[5];
  const float* lnb      = (const float*)d_in[6];
  float* out = (float*)d_out;

  char* p = (char*)d_ws;
  auto take = [&](size_t bytes) { void* r = (void*)p; p += (bytes + 255) & ~(size_t)255; return r; };
  float*    lnEf  = (float*)   take((size_t)gB * gE * gD * 4);
  __bf16*   lnEbf = (__bf16*)  take((size_t)gB * gE * gD * 2);
  __bf16*   hidbf = (__bf16*)  take((size_t)gB * gN * gD * 2);
  __bf16*   hidT  = (__bf16*)  take((size_t)gB * gD * gN * 2);
  __bf16*   nodek = (__bf16*)  take((size_t)gB * gN * gD * 2);
  __bf16*   ehbf  = (__bf16*)  take((size_t)gB * gE * gD * 2);
  __bf16*   ehT   = (__bf16*)  take((size_t)gB * gD * gE * 2);
  __bf16*   edgek = (__bf16*)  take((size_t)gB * gE * gD * 2);
  __bf16*   wnkT  = (__bf16*)  take((size_t)gD * gD * 2);
  __bf16*   wekT  = (__bf16*)  take((size_t)gD * gD * 2);
  unsigned* htb   = (unsigned*)take((size_t)gB * gE * gNW * 4);
  float*    m1    = (float*)   take((size_t)gB * gE * 4);
  float*    l1    = (float*)   take((size_t)gB * gE * 4);
  float*    m2    = (float*)   take((size_t)gB * gE * 4);
  float*    l2    = (float*)   take((size_t)gB * gE * 4);

  k_layernorm<<<gB * gE, gD, 0, stream>>>(edge_emb, lng, lnb, lnEf, lnEbf);
  k_cvt_hidden<<<(gB * gN * gD) / 256, 256, 0, stream>>>(hidden, hidbf, hidT);
  k_cvt_wT<<<(gD * gD) / 256, 256, 0, stream>>>(wnk, wnkT);
  k_cvt_wT<<<(gD * gD) / 256, 256, 0, stream>>>(wek, wekT);
  k_pack_ht<<<(gB * gE * gNW) / 256, 256, 0, stream>>>(ht, htb);

  k_gemm_bf<<<(gB * gN) / 16, 128, 0, stream>>>(hidbf, wnkT, nodek);      // node_k
  k_edge_stats<<<dim3(gE / 16, gB), 128, 0, stream>>>(lnEbf, nodek, htb, m1, l1);
  k_edge_h<<<dim3(gE / 16, gB), 128, 0, stream>>>(lnEbf, lnEf, nodek, hidT, htb, m1, l1, ehbf, ehT);
  k_gemm_bf<<<(gB * gE) / 16, 128, 0, stream>>>(ehbf, wekT, edgek);       // edge_k
  k_node_stats<<<dim3(gE / 16, gB), 128, 0, stream>>>(nodek, edgek, htb, m2, l2);
  k_node_out<<<dim3(gN / 16, gB), 128, 0, stream>>>(nodek, edgek, ehT, htb, m2, l2, out);
}
